// CausalSelfAttention_29549374996767
// MI455X (gfx1250) — matrix operand order repaired and checked by reading, not compiled, not run on hardware
//
#include <hip/hip_runtime.h>
#include <hip/hip_bf16.h>

typedef __attribute__((ext_vector_type(16))) _Float16 v16h;
typedef __attribute__((ext_vector_type(8)))  float    v8f;

#define T_SEQ  2048
#define NHEAD  16
#define DHEAD  64
#define DMODEL 1024
#define BATCH  2

union FragCvt { struct { uint4 lo, hi; } u; v16h h; };
union H8 { uint4 u; _Float16 h[8]; };
union H4 { uint2 u; _Float16 h[4]; };

__device__ __forceinline__ v16h make_frag(uint4 lo, uint4 hi) {
  FragCvt c; c.u.lo = lo; c.u.hi = hi; return c.h;
}

// async global -> LDS copy, 16B per lane; tracked by ASYNCcnt
__device__ __forceinline__ void async_copy_b128(void* lds_dst,
                                                const void* gsrc) {
  unsigned lds = (unsigned)(uintptr_t)lds_dst;   // flat addr truncates to LDS
  asm volatile("global_load_async_to_lds_b128 %0, %1, off"
               :: "v"(lds), "v"(gsrc) : "memory");
}

// ---------------------------------------------------------------------------
// fp32 -> f16 conversion (vectorized x4)
// ---------------------------------------------------------------------------
__global__ __launch_bounds__(256) void cvt_f32_f16(const float* __restrict__ in,
                                                   _Float16* __restrict__ out,
                                                   int n) {
  int i = (blockIdx.x * 256 + threadIdx.x) * 4;
  if (i + 3 < n) {
    float4 v = *(const float4*)(in + i);
    H4 o;
    o.h[0] = (_Float16)v.x; o.h[1] = (_Float16)v.y;
    o.h[2] = (_Float16)v.z; o.h[3] = (_Float16)v.w;
    *(uint2*)(out + i) = o.u;
  }
}

// ---------------------------------------------------------------------------
// NT GEMM: out[M,N] = A[M,K] * W[N,K]^T + bias, f16 operands, f32 accum.
// Block tile 128x128, BK=32, 8 waves; each wave: 2x4 WMMA tiles (32x64).
// Double-buffered LDS fed by async global->LDS copies (ASYNCcnt pipeline).
// MODE 0: f16 output scattered to [B,H,T,Dh]; MODE 1: f32 output [M,N].
// ---------------------------------------------------------------------------
template <int MODE>
__global__ __launch_bounds__(256) void gemm_nt_wmma(
    const _Float16* __restrict__ A, const _Float16* __restrict__ W,
    const float* __restrict__ bias, void* __restrict__ outPtr,
    float outScale) {
  __shared__ _Float16 As[2][128 * 40];
  __shared__ _Float16 Bs[2][128 * 40];
  const int t    = threadIdx.x;
  const int lane = t & 31;
  const int wid  = t >> 5;
  const int l16  = lane & 15;
  const int hi   = lane >> 4;         // which 16-lane half
  const int blockM = blockIdx.y * 128;
  const int blockN = blockIdx.x * 128;
  const int wm = (wid & 3) * 32;      // 4 waves along M
  const int wn = (wid >> 2) * 64;     // 2 waves along N

  // cooperative-load geometry: 512 chunks of 8 halves per 128x32 tile
  const int ldRow0 = (t + 0)   >> 2, ldCol0 = ((t + 0)   & 3) << 3;
  const int ldRow1 = (t + 256) >> 2, ldCol1 = ((t + 256) & 3) << 3;
  const _Float16* aRow0 = A + (size_t)(blockM + ldRow0) * DMODEL + ldCol0;
  const _Float16* aRow1 = A + (size_t)(blockM + ldRow1) * DMODEL + ldCol1;
  const _Float16* bRow0 = W + (size_t)(blockN + ldRow0) * DMODEL + ldCol0;
  const _Float16* bRow1 = W + (size_t)(blockN + ldRow1) * DMODEL + ldCol1;
  const int dstOff0 = ldRow0 * 40 + ldCol0;
  const int dstOff1 = ldRow1 * 40 + ldCol1;

  v8f acc[2][4] = {};

  // prologue: async-stage K-slab 0 into buffer 0 (4 ops per wave)
  async_copy_b128(&As[0][dstOff0], aRow0);
  async_copy_b128(&As[0][dstOff1], aRow1);
  async_copy_b128(&Bs[0][dstOff0], bRow0);
  async_copy_b128(&Bs[0][dstOff1], bRow1);

  int buf = 0;
  for (int k0 = 0; k0 < DMODEL; k0 += 32, buf ^= 1) {
    if (k0 + 32 < DMODEL) {
      // issue next slab into the other buffer, then wait only for the
      // current slab's 4 copies (async loads complete in issue order)
      const int nb = buf ^ 1;
      async_copy_b128(&As[nb][dstOff0], aRow0 + k0 + 32);
      async_copy_b128(&As[nb][dstOff1], aRow1 + k0 + 32);
      async_copy_b128(&Bs[nb][dstOff0], bRow0 + k0 + 32);
      async_copy_b128(&Bs[nb][dstOff1], bRow1 + k0 + 32);
      asm volatile("s_wait_asynccnt 0x4" ::: "memory");
    } else {
      asm volatile("s_wait_asynccnt 0x0" ::: "memory");
    }
    __syncthreads();

    v16h af[2], bf[4];
#pragma unroll
    for (int mt = 0; mt < 2; ++mt) {
      // A-operand 16x32: lanes0-15 hold K 0-7 & 16-23; lanes16-31 K 8-15 & 24-31
      const _Float16* p = &As[buf][(wm + mt * 16 + l16) * 40 + hi * 8];
      af[mt] = make_frag(*(const uint4*)p, *(const uint4*)(p + 16));
    }
#pragma unroll
    for (int nt = 0; nt < 4; ++nt) {
      // B-operand 32x16: lanes0-15 hold K 0-7 & 16-23; lanes16-31 K 8-15 & 24-31
      const _Float16* p = &Bs[buf][(wn + nt * 16 + l16) * 40 + hi * 8];
      bf[nt] = make_frag(*(const uint4*)p, *(const uint4*)(p + 16));
    }
#pragma unroll
    for (int mt = 0; mt < 2; ++mt)
#pragma unroll
      for (int nt = 0; nt < 4; ++nt)
        acc[mt][nt] = __builtin_amdgcn_wmma_f32_16x16x32_f16(
            false, af[mt], false, bf[nt], (short)0, acc[mt][nt], false, false);
    __syncthreads();
  }

  // ---------------- branch-free epilogues ----------------
  if (MODE == 0) {
    // f16 output scattered to [B,H,T,Dh].
    // 2048%128==0 -> batch index is block-uniform; (blockN+wn)%64==0 and
    // nt*16+l16 <= 63 -> head index is wave-uniform.
    const int bb = blockM >> 11;
    const int h  = (blockN + wn) >> 6;
    _Float16* op = (_Float16*)outPtr +
                   (((size_t)bb * NHEAD + h) * T_SEQ) * DHEAD;
    const int tt0 = (blockM & (T_SEQ - 1)) + wm + hi * 8;
#pragma unroll
    for (int nt = 0; nt < 4; ++nt) {
      const int d = nt * 16 + l16;
      const float b = bias[blockN + wn + d];
#pragma unroll
      for (int mt = 0; mt < 2; ++mt)
#pragma unroll
        for (int r = 0; r < 8; ++r) {
          const int tt = tt0 + mt * 16 + r;
          op[(size_t)tt * DHEAD + d] =
              (_Float16)((acc[mt][nt][r] + b) * outScale);
        }
    }
  } else {
    // f32 output, plain row-major [M, N]
    float* op = (float*)outPtr +
                (size_t)(blockM + wm + hi * 8) * DMODEL + blockN + wn;
#pragma unroll
    for (int nt = 0; nt < 4; ++nt) {
      const float b = bias[blockN + wn + nt * 16 + l16];
#pragma unroll
      for (int mt = 0; mt < 2; ++mt)
#pragma unroll
        for (int r = 0; r < 8; ++r)
          op[(size_t)(mt * 16 + r) * DMODEL + nt * 16 + l16] =
              (acc[mt][nt][r] + b) * outScale;
    }
  }
}

// ---------------------------------------------------------------------------
// Flash attention: one block = 64 query rows of one (b,h); 4 waves x 16 rows.
// Q pre-scaled by log2(e)/sqrt(Dh) so softmax uses exp2.
// ---------------------------------------------------------------------------
__global__ __launch_bounds__(128) void attn_flash_wmma(
    const _Float16* __restrict__ Qh, const _Float16* __restrict__ Kh,
    const _Float16* __restrict__ Vh, _Float16* __restrict__ Yh) {
  __shared__ _Float16 Vt[64 * 40];      // V tile transposed: [d][k]
  __shared__ _Float16 Ps[4][16 * 40];   // per-wave P staging: [q][k]
  const int t    = threadIdx.x;
  const int lane = t & 31;
  const int wid  = t >> 5;
  const int l16  = lane & 15;
  const int hi   = lane >> 4;
  const int bh   = blockIdx.x >> 5;     // 0..B*H-1
  const int qt   = blockIdx.x & 31;     // q tile of 64 rows
  const size_t base = (size_t)bh * T_SEQ * DHEAD;
  const int q0 = qt * 64 + wid * 16;

  // Q fragments (A-operand, d is contraction dim), kept in registers
  v16h aQ[2];
  {
    const _Float16* qp = Qh + base + (size_t)(q0 + l16) * DHEAD;
#pragma unroll
    for (int kd = 0; kd < 2; ++kd) {
      const _Float16* p = qp + kd * 32 + hi * 8;
      aQ[kd] = make_frag(*(const uint4*)p, *(const uint4*)(p + 16));
    }
  }

  float m_s[8], l_s[8];
  v8f accO[4] = {};
#pragma unroll
  for (int r = 0; r < 8; ++r) { m_s[r] = -1e30f; l_s[r] = 0.f; }

  const int nkb = qt * 2 + 2;           // 32-key blocks up to causal frontier
  for (int kb = 0; kb < nkb; ++kb) {
    const int kk = kb * 32;
    // prefetch next K/V tiles into cache (global_prefetch_b8)
    if (kb + 1 < nkb) {
      __builtin_prefetch(Kh + base + (size_t)(kk + 32 + lane) * DHEAD, 0, 3);
      __builtin_prefetch(Vh + base + (size_t)(kk + 32 + lane) * DHEAD, 0, 3);
    }
    __syncthreads();                    // protect previous Vt reads
    // stage V[32k x 64d] transposed into Vt[d][k]
#pragma unroll
    for (int c = 0; c < 2; ++c) {
      int ci   = t + c * 128;           // 0..255 chunks of 8 halves
      int krow = ci >> 3;
      int d0   = (ci & 7) << 3;
      H8 raw;
      raw.u = *(const uint4*)(Vh + base + (size_t)(kk + krow) * DHEAD + d0);
#pragma unroll
      for (int j = 0; j < 8; ++j) Vt[(d0 + j) * 40 + krow] = raw.h[j];
    }
    __syncthreads();

    // S = Q K^T   (contraction over d=64, two 16-key output tiles)
    v8f accS[2] = {};
#pragma unroll
    for (int kt = 0; kt < 2; ++kt) {
      const _Float16* kp =
          Kh + base + (size_t)(kk + kt * 16 + l16) * DHEAD + hi * 8;
#pragma unroll
      for (int ds = 0; ds < 2; ++ds) {
        v16h bK = make_frag(*(const uint4*)(kp + ds * 32),
                            *(const uint4*)(kp + ds * 32 + 16));
        accS[kt] = __builtin_amdgcn_wmma_f32_16x16x32_f16(
            false, aQ[ds], false, bK, (short)0, accS[kt], false, false);
      }
    }
    // causal mask
#pragma unroll
    for (int kt = 0; kt < 2; ++kt) {
      int key = kk + kt * 16 + l16;
#pragma unroll
      for (int r = 0; r < 8; ++r) {
        int q = q0 + r + hi * 8;
        if (key > q) accS[kt][r] = -1e30f;
      }
    }
    // row max across 32 keys (16 lanes per half hold one row)
    float rmax[8];
#pragma unroll
    for (int r = 0; r < 8; ++r) rmax[r] = fmaxf(accS[0][r], accS[1][r]);
#pragma unroll
    for (int off = 8; off >= 1; off >>= 1)
#pragma unroll
      for (int r = 0; r < 8; ++r)
        rmax[r] = fmaxf(rmax[r], __shfl_xor(rmax[r], off, 32));
    float sc[8];
#pragma unroll
    for (int r = 0; r < 8; ++r) {
      float mn = fmaxf(m_s[r], rmax[r]);
      sc[r] = exp2f(m_s[r] - mn);
      m_s[r] = mn;
    }
    // P = exp2(S - m); stage to per-wave LDS for layout conversion
    float rsum[8];
#pragma unroll
    for (int r = 0; r < 8; ++r) rsum[r] = 0.f;
#pragma unroll
    for (int kt = 0; kt < 2; ++kt)
#pragma unroll
      for (int r = 0; r < 8; ++r) {
        float p = exp2f(accS[kt][r] - m_s[r]);
        rsum[r] += p;
        Ps[wid][(r + hi * 8) * 40 + kt * 16 + l16] = (_Float16)p;
      }
#pragma unroll
    for (int off = 8; off >= 1; off >>= 1)
#pragma unroll
      for (int r = 0; r < 8; ++r)
        rsum[r] += __shfl_xor(rsum[r], off, 32);
#pragma unroll
    for (int r = 0; r < 8; ++r) l_s[r] = l_s[r] * sc[r] + rsum[r];
#pragma unroll
    for (int nt = 0; nt < 4; ++nt)
#pragma unroll
      for (int r = 0; r < 8; ++r) accO[nt][r] *= sc[r];

    // O += P(16x32) @ V(32x64); LDS ops from same wave are in-order
    v16h aP;
    {
      const _Float16* p = &Ps[wid][l16 * 40 + hi * 8];
      aP = make_frag(*(const uint4*)p, *(const uint4*)(p + 16));
    }
#pragma unroll
    for (int nt = 0; nt < 4; ++nt) {
      const _Float16* vp = &Vt[(nt * 16 + l16) * 40 + hi * 8];
      v16h bV = make_frag(*(const uint4*)vp, *(const uint4*)(vp + 16));
      accO[nt] = __builtin_amdgcn_wmma_f32_16x16x32_f16(
          false, aP, false, bV, (short)0, accO[nt], false, false);
    }
  }

  // epilogue: y in [B, T, C] (merge heads), f16 for the output projection
  const int bb = bh >> 4, h = bh & 15;
  _Float16* yp = Yh + ((size_t)bb * T_SEQ + q0 + hi * 8) * DMODEL + h * DHEAD;
#pragma unroll
  for (int r = 0; r < 8; ++r) {
    float inv = 1.0f / l_s[r];
#pragma unroll
    for (int nt = 0; nt < 4; ++nt)
      yp[(size_t)r * DMODEL + nt * 16 + l16] =
          (_Float16)(accO[nt][r] * inv);
  }
}

// ---------------------------------------------------------------------------
extern "C" void kernel_launch(void* const* d_in, const int* in_sizes, int n_in,
                              void* d_out, int out_size, void* d_ws,
                              size_t ws_size, hipStream_t stream) {
  (void)in_sizes; (void)n_in; (void)out_size; (void)ws_size;
  const float* x  = (const float*)d_in[0];
  const float* Wk = (const float*)d_in[1];
  const float* bk = (const float*)d_in[2];
  const float* Wq = (const float*)d_in[3];
  const float* bq = (const float*)d_in[4];
  const float* Wv = (const float*)d_in[5];
  const float* bv = (const float*)d_in[6];
  const float* Wp = (const float*)d_in[7];
  const float* bp = (const float*)d_in[8];
  float* out = (float*)d_out;

  char* ws = (char*)d_ws;
  const size_t MB = 1ull << 20;
  _Float16* xh  = (_Float16*)(ws + 0 * MB);   // 8 MB
  _Float16* wkh = (_Float16*)(ws + 8 * MB);   // 2 MB each
  _Float16* wqh = (_Float16*)(ws + 10 * MB);
  _Float16* wvh = (_Float16*)(ws + 12 * MB);
  _Float16* wph = (_Float16*)(ws + 14 * MB);
  _Float16* qh  = (_Float16*)(ws + 16 * MB);  // 8 MB each, [B,H,T,Dh]
  _Float16* kh  = (_Float16*)(ws + 24 * MB);
  _Float16* vh  = (_Float16*)(ws + 32 * MB);
  _Float16* yh  = (_Float16*)(ws + 40 * MB);  // 8 MB, [B,T,C]

  const int nX = BATCH * T_SEQ * DMODEL;      // 4M
  const int nW = DMODEL * DMODEL;             // 1M
  cvt_f32_f16<<<nX / 1024, 256, 0, stream>>>(x, xh, nX);
  cvt_f32_f16<<<nW / 1024, 256, 0, stream>>>(Wk, wkh, nW);
  cvt_f32_f16<<<nW / 1024, 256, 0, stream>>>(Wq, wqh, nW);
  cvt_f32_f16<<<nW / 1024, 256, 0, stream>>>(Wv, wvh, nW);
  cvt_f32_f16<<<nW / 1024, 256, 0, stream>>>(Wp, wph, nW);

  dim3 gg(DMODEL / 128, (BATCH * T_SEQ) / 128);     // (8, 32)
  const float qscale = 0.125f * 1.4426950408889634f; // (1/sqrt(64)) * log2(e)
  gemm_nt_wmma<0><<<gg, 256, 0, stream>>>(xh, wkh, bk, (void*)kh, 1.0f);
  gemm_nt_wmma<0><<<gg, 256, 0, stream>>>(xh, wqh, bq, (void*)qh, qscale);
  gemm_nt_wmma<0><<<gg, 256, 0, stream>>>(xh, wvh, bv, (void*)vh, 1.0f);

  attn_flash_wmma<<<BATCH * NHEAD * (T_SEQ / 64), 128, 0, stream>>>(qh, kh, vh, yh);

  gemm_nt_wmma<1><<<gg, 256, 0, stream>>>(yh, wph, bp, (void*)out, 1.0f);
}
